// MobilityGNN_53532472377746
// MI455X (gfx1250) — matrix operand order repaired and checked by reading, not compiled, not run on hardware
//
#include <hip/hip_runtime.h>

typedef __attribute__((ext_vector_type(16))) __bf16 v16bf;
typedef __attribute__((ext_vector_type(8)))  __bf16 bf16x8;
typedef __attribute__((ext_vector_type(8)))  float  v8f;
typedef unsigned int u32x4 __attribute__((ext_vector_type(4)));
typedef int          i32x4 __attribute__((ext_vector_type(4)));
typedef int          i32x8 __attribute__((ext_vector_type(8)));

#define NN_   8192
#define DIN_  256
#define DHID_ 512
#define DOUT_ 256

// ---------------------------------------------------------------- prep kernels

__global__ void colsum_k(const float* __restrict__ M, float* __restrict__ cs, int n) {
  int i = blockIdx.x * blockDim.x + threadIdx.x;
  float s = 0.f;
  for (int j = 0; j < n; ++j) s += M[(size_t)j * n + i];
  cs[i] = s;
}

// thr[i]  = 1e-6*(colsum+eps)   (mask: M[j,i] > thr[i]  <=>  A[j,i] > 1e-6)
// scale[i]= inv/(weight_sum+eps), hasE[i] = any mask
__global__ void colstats_k(const float* __restrict__ M, const float* __restrict__ cs,
                           float* __restrict__ thr, float* __restrict__ scale,
                           unsigned* __restrict__ hasE, int n) {
  int i = blockIdx.x * blockDim.x + threadIdx.x;
  float c = cs[i];
  float inv = 1.0f / (c + 1e-8f);
  float t = 1e-6f * (c + 1e-8f);
  float s = 0.f; unsigned any = 0u;
  for (int j = 0; j < n; ++j) {
    float m = M[(size_t)j * n + i];
    if (m > t) { s += m; any = 1u; }
  }
  float ws = s * inv;
  thr[i] = t;
  scale[i] = inv / (ws + 1e-8f);
  hasE[i] = any;
}

__global__ void f32_to_bf16_k(__bf16* __restrict__ d, const float* __restrict__ s, int total) {
  int idx = blockIdx.x * blockDim.x + threadIdx.x;
  if (idx < total) d[idx] = (__bf16)s[idx];
}

// dst[n*dstLD + colOff + k] = (bf16) src[k*Ncols + n]   (src is [K x Ncols] row-major)
__global__ void transpose_cvt_k(__bf16* __restrict__ dst, int dstLD, int colOff,
                                const float* __restrict__ src, int K, int Ncols) {
  int idx = blockIdx.x * blockDim.x + threadIdx.x;
  if (idx >= K * Ncols) return;
  int k = idx / Ncols, n = idx % Ncols;
  dst[(size_t)n * dstLD + colOff + k] = (__bf16)src[(size_t)k * Ncols + n];
}

__global__ void copy_cols_k(__bf16* __restrict__ dst, int dstLD, int colOff,
                            const __bf16* __restrict__ src, int srcLD, int total) {
  int idx = blockIdx.x * blockDim.x + threadIdx.x;
  if (idx >= total) return;
  int r = idx / srcLD, c = idx % srcLD;
  dst[(size_t)r * dstLD + colOff + c] = src[idx];
}

// ---------------------------------------------------------------- TDM helper
//
// DMA one [128 rows x 32 bf16] tile from global (row stride = strideElems bf16)
// into LDS with an 8-half pad per row -> row pitch 40 halfs (80B), matching the
// GEMM's LDS layout.  D# built per CDNA5 ISA 8.3-8.6.
__device__ inline void tdm_load_tile_2d(const void* gsrc, void* ldst, int strideElems) {
  unsigned lds = (unsigned)(size_t)ldst;  // low 32 bits of generic ptr = LDS offset
  unsigned long long ga = (unsigned long long)(size_t)gsrc;
  u32x4 g0;
  g0[0] = 1u;                                        // count=1 (valid user D#)
  g0[1] = lds;                                       // lds_addr
  g0[2] = (unsigned)(ga & 0xffffffffu);              // global_addr[31:0]
  g0[3] = (unsigned)((ga >> 32) & 0x01ffffffu)       // global_addr[56:32]
        | (2u << 30);                                // type = 2 ("image")
  unsigned td0 = (unsigned)strideElems;              // tensor_dim0 (>= tile_dim0)
  unsigned td1 = 1u << 20;                           // tensor_dim1 (large, tile in-bounds)
  i32x8 g1;
  g1[0] = (int)((1u << 16)                           // data_size = 2 bytes
              | (1u << 20)                           // pad_enable
              | (3u << 22)                           // pad_interval: 16 DWORDs (64B row)
              | (3u << 25));                         // pad_amount:   4 DWORDs (16B)
  g1[1] = (int)((td0 & 0xffffu) << 16);              // tensor_dim0[15:0]
  g1[2] = (int)((td0 >> 16) | ((td1 & 0xffffu) << 16)); // dim0[31:16] | dim1[15:0]
  g1[3] = (int)((td1 >> 16) | (32u << 16));          // dim1[31:16] | tile_dim0=32
  g1[4] = (int)(128u);                               // tile_dim1=128, tile_dim2=0
  g1[5] = (int)td0;                                  // tensor_dim0_stride[31:0]
  g1[6] = 0;                                         // stride hi + dim1_stride lo
  g1[7] = 0;
  i32x4 z4 = {0, 0, 0, 0};
#if __clang_major__ >= 23
  i32x8 z8 = {0, 0, 0, 0, 0, 0, 0, 0};
  __builtin_amdgcn_tensor_load_to_lds(g0, g1, z4, z4, z8, 0);
#else
  __builtin_amdgcn_tensor_load_to_lds(g0, g1, z4, z4, 0);
#endif
}

// ---------------------------------------------------------------- WMMA GEMM

struct GemmP {
  const __bf16* A; int lda;                       // AMODE 0: bf16 row-major A
  const float* Mf; int ldm; const float* thr;     // AMODE 1: A = mask(M)^T (on the fly)
  const __bf16* B; int ldb;                       // B given as [N][K] row-major (pre-transposed)
  int K;
  const float* bias0; const float* bias1;
  float* outF; int ldOutF;
  __bf16* outB; int ldOutB; int colOffB;
  const float* scale; const unsigned* hasE; const float* fallback; int ldFall;
};

__device__ inline v16bf combine16(bf16x8 lo, bf16x8 hi) {
  v16bf r;
#pragma unroll
  for (int e = 0; e < 8; ++e) { r[e] = lo[e]; r[e + 8] = hi[e]; }
  return r;
}

// Block: 256 threads (8 waves). Tile 128x128, BK=32. Wave tile 64x32 (4x2 frags).
// Double-buffered LDS: tile k+1 is DMA'd by the Tensor Data Mover (or fetched to
// registers for the masked-M path) while tile k is consumed by WMMA.  TENSORcnt
// is in-order per wave, so waiting "<= #ops just issued" proves tile k is done.
template <int AMODE, int EMODE>
__global__ __launch_bounds__(256) void gemm_bf16_wmma(GemmP p) {
  constexpr int LDS_LD = 40;       // 32 + 8 halfs pad (80B rows, conflict-free b128)
  constexpr int TILE = 128 * LDS_LD;
  __shared__ __bf16 sA[2 * TILE];
  __shared__ __bf16 sB[2 * TILE];
  __shared__ float sThr[128];

  const int tid = threadIdx.x;
  const int lane = tid & 31;
  const int wid = tid >> 5;
  const int wm = wid >> 2, wn = wid & 3;      // 2 x 4 wave grid
  const int m16 = lane & 15, lhalf = lane >> 4;
  const int rowBlk = blockIdx.y * 128;
  const int colBlk = blockIdx.x * 128;

  if constexpr (AMODE == 1) {
    if (tid < 128) sThr[tid] = p.thr[rowBlk + tid];
  }

  v8f acc[4][2] = {};
  const int nk = p.K >> 5;

  // pipeline registers for the masked-M path
  float mreg[16];
  const int jj = tid >> 3, ic = (tid & 7) << 4;

  // ---- prologue: start tile 0
  if constexpr (AMODE == 0) {
    if (wid == 0) {
      tdm_load_tile_2d(p.A + (size_t)rowBlk * p.lda, &sA[0], p.lda);
      tdm_load_tile_2d(p.B + (size_t)colBlk * p.ldb, &sB[0], p.ldb);
    }
  } else {
    if (wid == 0) tdm_load_tile_2d(p.B + (size_t)colBlk * p.ldb, &sB[0], p.ldb);
    const float* src = p.Mf + (size_t)jj * p.ldm + rowBlk + ic;
#pragma unroll
    for (int e = 0; e < 16; ++e) mreg[e] = src[e];
  }

  for (int kt = 0; kt < nk; ++kt) {
    const int cur = kt & 1, nxt = cur ^ 1;
    const int k0 = kt << 5;
    __syncthreads();   // all waves done reading buffer 'nxt' (tile kt-1)

    if constexpr (AMODE == 0) {
      if (wid == 0) {
        if (kt + 1 < nk) {
          tdm_load_tile_2d(p.A + (size_t)rowBlk * p.lda + k0 + 32, &sA[nxt * TILE], p.lda);
          tdm_load_tile_2d(p.B + (size_t)colBlk * p.ldb + k0 + 32, &sB[nxt * TILE], p.ldb);
          __builtin_amdgcn_s_wait_tensorcnt(2);   // tile kt complete (in-order)
        } else {
          __builtin_amdgcn_s_wait_tensorcnt(0);
        }
      }
    } else {
      if (wid == 0) {
        if (kt + 1 < nk)
          tdm_load_tile_2d(p.B + (size_t)colBlk * p.ldb + k0 + 32, &sB[nxt * TILE], p.ldb);
      }
      // transform pipelined M registers (tile kt): threshold + cvt + transpose
#pragma unroll
      for (int e = 0; e < 16; ++e) {
        float m = mreg[e];
        sA[cur * TILE + (ic + e) * LDS_LD + jj] = (__bf16)(m > sThr[ic + e] ? m : 0.0f);
      }
      // start fetching tile kt+1 of M into registers (clamped, branchless)
      {
        const int kn = (kt + 1 < nk) ? (k0 + 32) : 0;
        const float* src = p.Mf + (size_t)(kn + jj) * p.ldm + rowBlk + ic;
#pragma unroll
        for (int e = 0; e < 16; ++e) mreg[e] = src[e];
        __builtin_prefetch((const void*)(src + (size_t)32 * p.ldm), 0, 3);
      }
      if (wid == 0) {
        if (kt + 1 < nk) __builtin_amdgcn_s_wait_tensorcnt(1);  // B tile kt done
        else             __builtin_amdgcn_s_wait_tensorcnt(0);
      }
    }
    __syncthreads();   // buffer 'cur' ready for all waves

    const __bf16* A_ = &sA[cur * TILE];
    const __bf16* B_ = &sB[cur * TILE];

    // ---- fragments (ISA 7.12.2 16-bit layouts)
    v16bf af[4], bfr[2];
#pragma unroll
    for (int mi = 0; mi < 4; ++mi) {
      const int rr = wm * 64 + mi * 16 + m16;  // A: lane -> row M, halves split K
      bf16x8 lo = *(const bf16x8*)(&A_[rr * LDS_LD + lhalf * 8]);       // K = h*8 + 0..7
      bf16x8 hi = *(const bf16x8*)(&A_[rr * LDS_LD + lhalf * 8 + 16]);  // K = h*8 + 16..23
      af[mi] = combine16(lo, hi);
    }
#pragma unroll
    for (int ni = 0; ni < 2; ++ni) {
      const int cc = wn * 32 + ni * 16 + m16;  // B: lane -> col N, halves split K
      bf16x8 lo = *(const bf16x8*)(&B_[cc * LDS_LD + lhalf * 8]);       // K = h*8 + 0..7
      bf16x8 hi = *(const bf16x8*)(&B_[cc * LDS_LD + lhalf * 8 + 16]);  // K = h*8 + 16..23
      bfr[ni] = combine16(lo, hi);
    }
#pragma unroll
    for (int mi = 0; mi < 4; ++mi)
#pragma unroll
      for (int ni = 0; ni < 2; ++ni)
        acc[mi][ni] = __builtin_amdgcn_wmma_f32_16x16x32_bf16(
            false, af[mi], false, bfr[ni], (short)0, acc[mi][ni], false, false);
  }

  // ---- epilogue (C layout: acc[r] = C[h*8+r][lane&15])
#pragma unroll
  for (int mi = 0; mi < 4; ++mi) {
#pragma unroll
    for (int ni = 0; ni < 2; ++ni) {
      const int gc = colBlk + wn * 32 + ni * 16 + m16;
      const int grb = rowBlk + wm * 64 + mi * 16 + lhalf * 8;
#pragma unroll
      for (int r = 0; r < 8; ++r) {
        const int gr = grb + r;
        float v = acc[mi][ni][r];
        if constexpr (EMODE == 1) {          // T = h@W_in + b_in ; also bf16 T^T
          v += p.bias0[gc];
          p.outF[(size_t)gr * p.ldOutF + gc] = v;
          p.outB[(size_t)gc * p.ldOutB + gr] = (__bf16)v;
        } else if constexpr (EMODE == 2) {   // agg = hasE ? scale*P : T
          float a = p.hasE[gr] ? v * p.scale[gr]
                               : p.fallback[(size_t)gr * p.ldFall + gc];
          p.outB[(size_t)gr * p.ldOutB + p.colOffB + gc] = (__bf16)a;
        } else {                             // pre-LN: + b_out + b_self
          v += p.bias0[gc] + p.bias1[gc];
          p.outF[(size_t)gr * p.ldOutF + gc] = v;
        }
      }
    }
  }
}

// ---------------------------------------------------------------- LayerNorm+ReLU

__global__ __launch_bounds__(256) void layernorm_relu_k(const float* __restrict__ X,
                                                        const float* __restrict__ g,
                                                        const float* __restrict__ be,
                                                        __bf16* outB, float* outF, int D) {
  __shared__ float rs[8], rss[8];
  const int row = blockIdx.x, tid = threadIdx.x;
  const int lane = tid & 31, wid = tid >> 5;
  const float* x = X + (size_t)row * D;
  float s = 0.f, ss = 0.f;
  for (int n = tid; n < D; n += 256) { float v = x[n]; s += v; ss += v * v; }
  for (int o = 16; o > 0; o >>= 1) { s += __shfl_xor(s, o); ss += __shfl_xor(ss, o); }
  if (lane == 0) { rs[wid] = s; rss[wid] = ss; }
  __syncthreads();
  s = 0.f; ss = 0.f;
  for (int w2 = 0; w2 < 8; ++w2) { s += rs[w2]; ss += rss[w2]; }
  float mu = s / D;
  float var = ss / D - mu * mu;
  float rstd = rsqrtf(var + 1e-5f);
  for (int n = tid; n < D; n += 256) {
    float v = (x[n] - mu) * rstd * g[n] + be[n];
    v = fmaxf(v, 0.f);
    if (outB) outB[(size_t)row * D + n] = (__bf16)v;
    if (outF) outF[(size_t)row * D + n] = v;
  }
}

// ---------------------------------------------------------------- host

extern "C" void kernel_launch(void* const* d_in, const int* in_sizes, int n_in,
                              void* d_out, int out_size, void* d_ws, size_t ws_size,
                              hipStream_t stream) {
  (void)in_sizes; (void)n_in; (void)out_size; (void)ws_size;
  const float* h0      = (const float*)d_in[0];
  const float* M       = (const float*)d_in[1];
  const float* W_in1   = (const float*)d_in[2];
  const float* b_in1   = (const float*)d_in[3];
  const float* W_out1  = (const float*)d_in[4];
  const float* b_out1  = (const float*)d_in[5];
  const float* W_self1 = (const float*)d_in[6];
  const float* b_self1 = (const float*)d_in[7];
  const float* g1      = (const float*)d_in[8];
  const float* be1     = (const float*)d_in[9];
  const float* W_in2   = (const float*)d_in[10];
  const float* b_in2   = (const float*)d_in[11];
  const float* W_out2  = (const float*)d_in[12];
  const float* b_out2  = (const float*)d_in[13];
  const float* W_self2 = (const float*)d_in[14];
  const float* b_self2 = (const float*)d_in[15];
  const float* g2      = (const float*)d_in[16];
  const float* be2     = (const float*)d_in[17];

  char* w = (char*)d_ws;
  auto bump = [&](size_t bytes) -> void* {
    void* pp = (void*)w;
    w += (bytes + 255) & ~(size_t)255;
    return pp;
  };
  float*    csum  = (float*)bump((size_t)NN_ * 4);
  float*    thr   = (float*)bump((size_t)NN_ * 4);
  float*    scale = (float*)bump((size_t)NN_ * 4);
  unsigned* hasE  = (unsigned*)bump((size_t)NN_ * 4);
  float*    Tbuf  = (float*)bump((size_t)NN_ * DHID_ * 4);
  __bf16*   Tbt   = (__bf16*)bump((size_t)DHID_ * NN_ * 2);
  __bf16*   ABc   = (__bf16*)bump((size_t)NN_ * 768 * 2);
  float*    Cbuf  = (float*)bump((size_t)NN_ * DHID_ * 4);
  __bf16*   hbf   = (__bf16*)bump((size_t)NN_ * DHID_ * 2);
  __bf16*   Win1t = (__bf16*)bump((size_t)DHID_ * DIN_ * 2);
  __bf16*   Wc1t  = (__bf16*)bump((size_t)DHID_ * 768 * 2);
  __bf16*   Win2t = (__bf16*)bump((size_t)DOUT_ * DHID_ * 2);
  __bf16*   Wc2t  = (__bf16*)bump((size_t)DOUT_ * 768 * 2);

  // column stats of M (shared by both layers)
  colsum_k<<<NN_ / 256, 256, 0, stream>>>(M, csum, NN_);
  colstats_k<<<NN_ / 256, 256, 0, stream>>>(M, csum, thr, scale, hasE, NN_);
  // bf16 inputs / transposed bf16 weights
  f32_to_bf16_k<<<(NN_ * DIN_) / 256, 256, 0, stream>>>(hbf, h0, NN_ * DIN_);
  transpose_cvt_k<<<(DIN_ * DHID_) / 256, 256, 0, stream>>>(Win1t, DIN_, 0, W_in1, DIN_, DHID_);
  transpose_cvt_k<<<(DHID_ * DHID_) / 256, 256, 0, stream>>>(Wc1t, 768, 0, W_out1, DHID_, DHID_);
  transpose_cvt_k<<<(DIN_ * DHID_) / 256, 256, 0, stream>>>(Wc1t, 768, DHID_, W_self1, DIN_, DHID_);
  transpose_cvt_k<<<(DHID_ * DOUT_) / 256, 256, 0, stream>>>(Win2t, DHID_, 0, W_in2, DHID_, DOUT_);
  transpose_cvt_k<<<(DOUT_ * DOUT_) / 256, 256, 0, stream>>>(Wc2t, 768, 0, W_out2, DOUT_, DOUT_);
  transpose_cvt_k<<<(DHID_ * DOUT_) / 256, 256, 0, stream>>>(Wc2t, 768, DOUT_, W_self2, DHID_, DOUT_);

  // ================= layer 1 : din=256, dh=512 =================
  {
    GemmP p{}; p.A = hbf; p.lda = DIN_; p.B = Win1t; p.ldb = DIN_; p.K = DIN_;
    p.bias0 = b_in1; p.outF = Tbuf; p.ldOutF = DHID_; p.outB = Tbt; p.ldOutB = NN_;
    gemm_bf16_wmma<0, 1><<<dim3(DHID_ / 128, NN_ / 128), 256, 0, stream>>>(p);
  }
  copy_cols_k<<<(NN_ * DIN_) / 256, 256, 0, stream>>>(ABc, 768, DHID_, hbf, DIN_, NN_ * DIN_);
  {
    GemmP p{}; p.Mf = M; p.ldm = NN_; p.thr = thr; p.B = Tbt; p.ldb = NN_; p.K = NN_;
    p.scale = scale; p.hasE = hasE; p.fallback = Tbuf; p.ldFall = DHID_;
    p.outB = ABc; p.ldOutB = 768; p.colOffB = 0;
    gemm_bf16_wmma<1, 2><<<dim3(DHID_ / 128, NN_ / 128), 256, 0, stream>>>(p);
  }
  {
    GemmP p{}; p.A = ABc; p.lda = 768; p.B = Wc1t; p.ldb = 768; p.K = 768;
    p.bias0 = b_out1; p.bias1 = b_self1; p.outF = Cbuf; p.ldOutF = DHID_;
    gemm_bf16_wmma<0, 3><<<dim3(DHID_ / 128, NN_ / 128), 256, 0, stream>>>(p);
  }
  layernorm_relu_k<<<NN_, 256, 0, stream>>>(Cbuf, g1, be1, hbf, nullptr, DHID_);

  // ================= layer 2 : din=512, dh=256 =================
  {
    GemmP p{}; p.A = hbf; p.lda = DHID_; p.B = Win2t; p.ldb = DHID_; p.K = DHID_;
    p.bias0 = b_in2; p.outF = Tbuf; p.ldOutF = DOUT_; p.outB = Tbt; p.ldOutB = NN_;
    gemm_bf16_wmma<0, 1><<<dim3(DOUT_ / 128, NN_ / 128), 256, 0, stream>>>(p);
  }
  copy_cols_k<<<(NN_ * DHID_) / 256, 256, 0, stream>>>(ABc, 768, DOUT_, hbf, DHID_, NN_ * DHID_);
  {
    GemmP p{}; p.Mf = M; p.ldm = NN_; p.thr = thr; p.B = Tbt; p.ldb = NN_; p.K = NN_;
    p.scale = scale; p.hasE = hasE; p.fallback = Tbuf; p.ldFall = DOUT_;
    p.outB = ABc; p.ldOutB = 768; p.colOffB = 0;
    gemm_bf16_wmma<1, 2><<<dim3(DOUT_ / 128, NN_ / 128), 256, 0, stream>>>(p);
  }
  {
    GemmP p{}; p.A = ABc; p.lda = 768; p.B = Wc2t; p.ldb = 768; p.K = 768;
    p.bias0 = b_out2; p.bias1 = b_self2; p.outF = Cbuf; p.ldOutF = DOUT_;
    gemm_bf16_wmma<0, 3><<<dim3(DOUT_ / 128, NN_ / 128), 256, 0, stream>>>(p);
  }
  layernorm_relu_k<<<NN_, 256, 0, stream>>>(Cbuf, g2, be2, nullptr, (float*)d_out, DOUT_);
}
